// LSTM_65635690218199
// MI455X (gfx1250) — compile-verified, matrix operand order checked by reading, not run on hardware
//
#include <hip/hip_runtime.h>

// ---------------------------------------------------------------------------
// MI455X fused 2-layer LSTM (B=256, T=1024, H=512, I=1).
//  - 16 persistent workgroups; each owns a 16-row batch tile end-to-end
//    (both layers, all 1024 steps): zero inter-WG traffic, recurrence on-chip.
//  - 512 threads = 16 wave32s per WG (4 waves/SIMD) to hide L2 weight-stream
//    latency under the WMMA chains; each wave owns 32 of the 512 h-columns.
//  - h1/h2 state: bf16 in LDS, double buffered (64 KB). c1/c2: registers
//    (each cell owned by exactly one lane/VGPR slot of the WMMA C layout).
//  - Weights: bf16 in workspace, streamed from L2 (6 MB << 192 MB L2).
//  - Math: v_wmma_f32_16x16x32_bf16, f32 accumulate (f32 WMMA is 8x less K
//    per instruction; fp8 would drift over a 1024-step recurrence).
// ---------------------------------------------------------------------------

typedef __attribute__((ext_vector_type(16))) __bf16 v16bf;
typedef __attribute__((ext_vector_type(8)))  float  v8f;

#define HID     512
#define FOURH   2048
#define TSTEPS  1024
#define BTILE   16
#define NTHREADS 512
#define NWAVES   16      // wave owns 512/16 = 32 h-cols = 2 tiles of 16
#define CT       2
#define LOG2E   1.4426950408889634f

__device__ __forceinline__ float fsig(float x) {
    float e = __builtin_amdgcn_exp2f(-x * LOG2E);
    return __builtin_amdgcn_rcpf(1.0f + e);
}
__device__ __forceinline__ float ftanh(float x) {
    float e = __builtin_amdgcn_exp2f(-2.0f * x * LOG2E);
    return 1.0f - 2.0f * __builtin_amdgcn_rcpf(1.0f + e);
}

__global__ void cvt_bf16_kernel(const float* __restrict__ s,
                                __bf16* __restrict__ d, int n) {
    int i = blockIdx.x * blockDim.x + threadIdx.x;
    if (i < n) d[i] = (__bf16)s[i];
}

__global__ __launch_bounds__(NTHREADS)
void lstm2_fused_kernel(const float* __restrict__ x,      // [256,1024,1]
                        const float* __restrict__ Wih1,   // [2048] (I=1)
                        const float* __restrict__ bih1,
                        const float* __restrict__ bhh1,
                        const float* __restrict__ bih2,
                        const float* __restrict__ bhh2,
                        const __bf16* __restrict__ Whh1,  // [2048,512] bf16
                        const __bf16* __restrict__ Wih2,  // [2048,512] bf16
                        const __bf16* __restrict__ Whh2,  // [2048,512] bf16
                        float* __restrict__ out)          // [256,512]
{
    __shared__ __bf16 h1b[2][BTILE * HID];   // 32 KB
    __shared__ __bf16 h2b[2][BTILE * HID];   // 32 KB

    const int tid  = threadIdx.x;
    const int lane = tid & 31;
    const int wv   = tid >> 5;        // wave 0..15; owns h-cols [wv*32, wv*32+32)
    const int nl   = lane & 15;       // A-frag row m / B-frag col n / C col n
    const int kh   = lane >> 4;       // K-half select for A/B frags; C row-half
    const int b0   = blockIdx.x * BTILE;

    // zero initial h state (both buffers)
    for (int i = tid; i < BTILE * HID; i += NTHREADS) {
        h1b[0][i] = (__bf16)0.f; h1b[1][i] = (__bf16)0.f;
        h2b[0][i] = (__bf16)0.f; h2b[1][i] = (__bf16)0.f;
    }

    // t-invariant per-(ct,gate) scalars hoisted out of the recurrence
    float w1s[CT][4], b1s[CT][4], b2s[CT][4];
    #pragma unroll
    for (int ct = 0; ct < CT; ++ct)
        #pragma unroll
        for (int gate = 0; gate < 4; ++gate) {
            const int gcol = gate * HID + wv * (CT * 16) + ct * 16 + nl;
            w1s[ct][gate] = Wih1[gcol];
            b1s[ct][gate] = bih1[gcol] + bhh1[gcol];
            b2s[ct][gate] = bih2[gcol] + bhh2[gcol];
        }

    // c state in registers: cell (m = r + 8*kh, n = wv*32 + ct*16 + nl)
    float c1r[CT][8], c2r[CT][8];
    #pragma unroll
    for (int ct = 0; ct < CT; ++ct)
        #pragma unroll
        for (int r = 0; r < 8; ++r) { c1r[ct][r] = 0.f; c2r[ct][r] = 0.f; }
    __syncthreads();

    int p = 0;
    for (int t = 0; t < TSTEPS; ++t) {
        const __bf16* h1r = h1b[p];
        __bf16*       h1w = h1b[p ^ 1];
        const __bf16* h2r = h2b[p];
        __bf16*       h2w = h2b[p ^ 1];

        // x[b, t] for the 8 batch rows this lane's C-slots cover
        float xv[8];
        #pragma unroll
        for (int r = 0; r < 8; ++r)
            xv[r] = x[(size_t)(b0 + r + 8 * kh) * TSTEPS + t];

        // ------------- layer 1: g = h1 @ Whh1^T + x*Wih1 + (bih1+bhh1) -------------
        #pragma unroll
        for (int ct = 0; ct < CT; ++ct) {
            const int n0 = wv * (CT * 16) + ct * 16;
            v8f gacc[4];
            #pragma unroll
            for (int gate = 0; gate < 4; ++gate) {
                const int gcol = gate * HID + n0 + nl;
                const float w1v = w1s[ct][gate];
                const float bv  = b1s[ct][gate];
                v8f acc;
                #pragma unroll
                for (int r = 0; r < 8; ++r) acc[r] = xv[r] * w1v + bv;

                const __bf16* ap = h1r  + nl * HID + kh * 16;           // A: LDS
                const __bf16* bp = Whh1 + (size_t)gcol * HID + kh * 16; // B: L2
                #pragma unroll 4
                for (int k = 0; k < HID; k += 32) {
                    v16bf a  = *(const v16bf*)(ap + k);
                    v16bf bm = *(const v16bf*)(bp + k);
                    acc = __builtin_amdgcn_wmma_f32_16x16x32_bf16(
                              false, a, false, bm, (short)0, acc, false, false);
                }
                gacc[gate] = acc;
            }
            // cell update (each (r,lane) slot owns its (m,n) cell)
            #pragma unroll
            for (int r = 0; r < 8; ++r) {
                const int m = r + 8 * kh;
                const int n = n0 + nl;
                float iv = fsig(gacc[0][r]);
                float fv = fsig(gacc[1][r]);
                float gv = ftanh(gacc[2][r]);
                float ov = fsig(gacc[3][r]);
                float cn = fv * c1r[ct][r] + iv * gv;
                c1r[ct][r] = cn;
                h1w[m * HID + n] = (__bf16)(ov * ftanh(cn));
            }
        }
        __syncthreads();   // h1w complete before layer 2 consumes it

        // --------- layer 2: g = h1_new @ Wih2^T + h2 @ Whh2^T + (bih2+bhh2) ---------
        #pragma unroll
        for (int ct = 0; ct < CT; ++ct) {
            const int n0 = wv * (CT * 16) + ct * 16;
            v8f gacc[4];
            #pragma unroll
            for (int gate = 0; gate < 4; ++gate) {
                const int gcol = gate * HID + n0 + nl;
                v8f acc;
                #pragma unroll
                for (int r = 0; r < 8; ++r) acc[r] = b2s[ct][gate];

                // input GEMM: A = fresh h1 (LDS), B = Wih2 row gcol
                const __bf16* ap = h1w  + nl * HID + kh * 16;
                const __bf16* bp = Wih2 + (size_t)gcol * HID + kh * 16;
                #pragma unroll 4
                for (int k = 0; k < HID; k += 32) {
                    v16bf a  = *(const v16bf*)(ap + k);
                    v16bf bm = *(const v16bf*)(bp + k);
                    acc = __builtin_amdgcn_wmma_f32_16x16x32_bf16(
                              false, a, false, bm, (short)0, acc, false, false);
                }
                // recurrence GEMM: A = previous h2 (LDS), B = Whh2 row gcol
                const __bf16* ap2 = h2r  + nl * HID + kh * 16;
                const __bf16* bp2 = Whh2 + (size_t)gcol * HID + kh * 16;
                #pragma unroll 4
                for (int k = 0; k < HID; k += 32) {
                    v16bf a  = *(const v16bf*)(ap2 + k);
                    v16bf bm = *(const v16bf*)(bp2 + k);
                    acc = __builtin_amdgcn_wmma_f32_16x16x32_bf16(
                              false, a, false, bm, (short)0, acc, false, false);
                }
                gacc[gate] = acc;
            }
            #pragma unroll
            for (int r = 0; r < 8; ++r) {
                const int m = r + 8 * kh;
                const int n = n0 + nl;
                float iv = fsig(gacc[0][r]);
                float fv = fsig(gacc[1][r]);
                float gv = ftanh(gacc[2][r]);
                float ov = fsig(gacc[3][r]);
                float cn = fv * c2r[ct][r] + iv * gv;
                c2r[ct][r] = cn;
                float hv = ov * ftanh(cn);
                h2w[m * HID + n] = (__bf16)hv;
                if (t == TSTEPS - 1)
                    out[(size_t)(b0 + m) * HID + n] = hv;  // final h2, f32
            }
        }
        __syncthreads();   // h2w complete before next step reads it
        p ^= 1;
    }
}

extern "C" void kernel_launch(void* const* d_in, const int* in_sizes, int n_in,
                              void* d_out, int out_size, void* d_ws, size_t ws_size,
                              hipStream_t stream) {
    // setup_inputs order:
    // 0 input_seq [256,1024,1]  1 Wih1 [2048,1]  2 Whh1 [2048,512]
    // 3 bih1 [2048]  4 bhh1 [2048]  5 Wih2 [2048,512]  6 Whh2 [2048,512]
    // 7 bih2 [2048]  8 bhh2 [2048]
    const float* x    = (const float*)d_in[0];
    const float* Wih1 = (const float*)d_in[1];
    const float* Whh1 = (const float*)d_in[2];
    const float* bih1 = (const float*)d_in[3];
    const float* bhh1 = (const float*)d_in[4];
    const float* Wih2 = (const float*)d_in[5];
    const float* Whh2 = (const float*)d_in[6];
    const float* bih2 = (const float*)d_in[7];
    const float* bhh2 = (const float*)d_in[8];
    float* out = (float*)d_out;

    const int NW = FOURH * HID;              // 1,048,576 elements per matrix
    __bf16* w1bf  = (__bf16*)d_ws;           // Whh1 bf16 (2 MB)
    __bf16* wi2bf = w1bf  + NW;              // Wih2 bf16 (2 MB)
    __bf16* wh2bf = wi2bf + NW;              // Whh2 bf16 (2 MB)

    const int cthr = 256;
    const int cblk = (NW + cthr - 1) / cthr;
    cvt_bf16_kernel<<<cblk, cthr, 0, stream>>>(Whh1, w1bf,  NW);
    cvt_bf16_kernel<<<cblk, cthr, 0, stream>>>(Wih2, wi2bf, NW);
    cvt_bf16_kernel<<<cblk, cthr, 0, stream>>>(Whh2, wh2bf, NW);

    // 16 workgroups x 512 threads (16 wave32s): each WG owns a 16-row batch tile.
    lstm2_fused_kernel<<<256 / BTILE, NTHREADS, 0, stream>>>(
        x, Wih1, bih1, bhh1, bih2, bhh2, w1bf, wi2bf, wh2bf, out);
}
